// PagedAttention_55559696941695
// MI455X (gfx1250) — matrix operand order repaired and checked by reading, not compiled, not run on hardware
//
#include <hip/hip_runtime.h>
#include <cstdint>
#include <cstddef>

// ---------------- problem constants ----------------
constexpr int T_   = 2048;
constexpr int HID_ = 4096;
constexpr int NH_  = 32;
constexpr int KVH_ = 8;
constexpr int D_   = 128;
constexpr float SCALE_ = 0.088388347648318447f;   // D^-0.5
constexpr float NEG_BIG = -1e30f;

// ---------------- WMMA types ----------------
typedef __attribute__((ext_vector_type(16))) __bf16 v16bf;
typedef __attribute__((ext_vector_type(8)))  float  v8f;
typedef __attribute__((ext_vector_type(4)))  unsigned int u32x4;
typedef __attribute__((ext_vector_type(8)))  unsigned int u32x8;

union FragB {
    uint4 u[2];
    v16bf v;
    unsigned short s[16];
};

static __device__ __forceinline__ v8f zero8() {
    v8f z;
#pragma unroll
    for (int i = 0; i < 8; ++i) z[i] = 0.0f;
    return z;
}

static __device__ __forceinline__ v8f wmma_bf16(const FragB& a, const FragB& b, v8f c) {
    return __builtin_amdgcn_wmma_f32_16x16x32_bf16(false, a.v, false, b.v,
                                                   (short)0, c, false, false);
}

// hardware f32 -> bf16 (RNE) via native conversion
static __device__ __forceinline__ unsigned short f2bf(float f) {
    __bf16 h = (__bf16)f;
    return __builtin_bit_cast(unsigned short, h);
}

static __device__ __forceinline__ float rmax16(float x) {
#pragma unroll
    for (int m = 1; m < 16; m <<= 1) x = fmaxf(x, __shfl_xor(x, m, 32));
    return x;
}
static __device__ __forceinline__ float rsum16(float x) {
#pragma unroll
    for (int m = 1; m < 16; m <<= 1) x += __shfl_xor(x, m, 32);
    return x;
}

static __device__ __forceinline__ void lds_wait() {
#if __has_builtin(__builtin_amdgcn_s_wait_dscnt)
    __builtin_amdgcn_s_wait_dscnt(0);
#else
    asm volatile("s_wait_dscnt 0x0" ::: "memory");
#endif
}

// ---------------- Tensor Data Mover: 2D tile -> LDS ----------------
// D# per CDNA5 ISA. Issued once per wave
// (EXEC ignored); tracked with TENSORcnt.
static __device__ __forceinline__ void tdm_load_2d(
    unsigned lds_addr,              // byte address in LDS
    const void* gptr,               // global byte address of tile start
    unsigned g1w0,                  // data_size + pad config word
    unsigned tensor_d0, unsigned tensor_d1,
    unsigned tile_d0, unsigned tile_d1,
    unsigned long long stride0)     // dim0 stride in elements
{
    unsigned long long ga = (unsigned long long)(uintptr_t)gptr;
    u32x4 g0;
    g0[0] = 1u;                                   // count=1, user descriptor
    g0[1] = lds_addr;                             // lds_addr
    g0[2] = (unsigned)ga;                         // global_addr[31:0]
    g0[3] = ((unsigned)(ga >> 32) & 0x01FFFFFFu)  // global_addr[56:32]
            | 0x80000000u;                        // type=2 ("image")
    u32x8 g1;
    g1[0] = g1w0;                                 // mask=0 | data_size | pad cfg
    g1[1] = (tensor_d0 & 0xFFFFu) << 16;          // tensor_dim0[15:0] (atomic addr=0)
    g1[2] = ((tensor_d0 >> 16) & 0xFFFFu) | ((tensor_d1 & 0xFFFFu) << 16);
    g1[3] = ((tensor_d1 >> 16) & 0xFFFFu) | ((tile_d0 & 0xFFFFu) << 16);
    g1[4] = tile_d1 & 0xFFFFu;                    // tile_dim1, tile_dim2=0 (2D)
    g1[5] = (unsigned)(stride0 & 0xFFFFFFFFull);  // tensor_dim0_stride[31:0]
    g1[6] = (unsigned)((stride0 >> 32) & 0xFFFFull); // stride0[47:32], stride1=0
    g1[7] = 0u;
    asm volatile("tensor_load_to_lds %0, %1" :: "s"(g0), "s"(g1) : "memory");
}

static __device__ __forceinline__ void tdm_wait() {
    __builtin_amdgcn_s_wait_tensorcnt(0);
}

// data_size=2B (bit16), pad_enable (bit20), pad_interval (bits 24:22), pad_amount (bits 31:25)
// GEMM A tile: rows of 32 bf16 = 16 DW (code 3), pad 4 DW (code 3) -> 80B stride
constexpr unsigned TDM_W0_GEMM = (1u << 16) | (1u << 20) | (3u << 22) | (3u << 25);
// Flash K tile: rows of 128 bf16 = 64 DW (code 5), pad 4 DW (code 3) -> 272B stride
constexpr unsigned TDM_W0_K    = (1u << 16) | (1u << 20) | (5u << 22) | (3u << 25);

// ---------------- elementwise kernels ----------------
__global__ void cvt_f32_bf16(const float* __restrict__ in,
                             unsigned short* __restrict__ out, int n) {
    int i = blockIdx.x * blockDim.x + threadIdx.x;
    if (i < n) out[i] = f2bf(in[i]);
}

// RoPE: src fp32 (T x nheads*128) -> dst bf16, half = 64
__global__ void rope_kernel(const float* __restrict__ src,
                            unsigned short* __restrict__ dst,
                            const int* __restrict__ pos, int nheads, int total) {
    int idx = blockIdx.x * blockDim.x + threadIdx.x;   // t*nheads*64 + h*64 + i
    if (idx >= total) return;
    int i    = idx & 63;
    int rest = idx >> 6;
    int hh   = rest % nheads;
    int t    = rest / nheads;
    float p    = (float)pos[t];
    float invf = __expf(-(float)i * (9.2103403720f / 64.0f));   // theta^-(i/64)
    float fr   = p * invf;
    float s, c;
    __sincosf(fr, &s, &c);
    size_t b = (size_t)t * nheads * 128 + (size_t)hh * 128;
    float v1 = src[b + i];
    float v2 = src[b + i + 64];
    dst[b + i]      = f2bf(v1 * c - v2 * s);
    dst[b + i + 64] = f2bf(v2 * c + v1 * s);
}

// tiled transpose + cast: src f32 [rows][cols] -> dst bf16 [cols][rows]
__global__ __launch_bounds__(256)
void transpose_cast(const float* __restrict__ src, unsigned short* __restrict__ dst,
                    int rows, int cols) {
    __shared__ float tile[32][33];
    int c0 = blockIdx.x * 32;
    int r0 = blockIdx.y * 32;
    int tx = threadIdx.x;        // 0..31
    int ty = threadIdx.y;        // 0..7
#pragma unroll
    for (int j = 0; j < 32; j += 8)
        tile[ty + j][tx] = src[(size_t)(r0 + ty + j) * cols + c0 + tx];
    __syncthreads();
#pragma unroll
    for (int j = 0; j < 32; j += 8)
        dst[(size_t)(c0 + ty + j) * rows + r0 + tx] = f2bf(tile[tx][ty + j]);
}

// ---------------- GEMM: C(MxN,f32) = A(MxK,bf16) @ B(NxK,f32)^T ----------------
// block tile 128x128, BK=32, 256 threads (8 waves, 2x4), wave tile 64x32.
// A tile staged by TDM (tensor_load_to_lds); B tile staged + converted by VALU.
__global__ __launch_bounds__(256)
void gemm_bf16_f32w(const unsigned short* __restrict__ A,
                    const float* __restrict__ B,
                    float* __restrict__ C, int M, int N, int K) {
    __shared__ __align__(16) unsigned short As[128 * 40];
    __shared__ __align__(16) unsigned short Bs[128 * 40];

    const int tid  = threadIdx.x;
    const int lane = tid & 31;
    const int wid  = tid >> 5;
    const int wm   = wid >> 2;          // 0..1
    const int wn   = wid & 3;           // 0..3
    const int m0   = blockIdx.y * 128;
    const int n0   = blockIdx.x * 128;
    const int ln   = lane & 15;
    const int half = lane >> 4;
    const int khalf = half * 8;

    const int lrow = tid >> 1;          // 0..127
    const int lcol = (tid & 1) * 16;    // 0 or 16

    const unsigned asAddr = (unsigned)(uintptr_t)(&As[0]);

    v8f acc[4][2];
#pragma unroll
    for (int i = 0; i < 4; ++i)
#pragma unroll
        for (int j = 0; j < 2; ++j) acc[i][j] = zero8();

    for (int kb = 0; kb < K; kb += 32) {
        __syncthreads();
        if (wid == 0) {   // A tile 128 rows x 32 bf16, row stride K elems -> As (stride 40)
            tdm_load_2d(asAddr, A + (size_t)m0 * K + kb, TDM_W0_GEMM,
                        32, 128, 32, 128, (unsigned long long)K);
        }
        {   // stage B tile (f32 -> bf16) on all waves, overlapping the TDM DMA
            const float4* src = (const float4*)(B + (size_t)(n0 + lrow) * K + kb + lcol);
            unsigned short tmp[16];
#pragma unroll
            for (int i = 0; i < 4; ++i) {
                float4 f = src[i];
                tmp[i * 4 + 0] = f2bf(f.x);
                tmp[i * 4 + 1] = f2bf(f.y);
                tmp[i * 4 + 2] = f2bf(f.z);
                tmp[i * 4 + 3] = f2bf(f.w);
            }
            uint4* dst = (uint4*)(Bs + lrow * 40 + lcol);
            dst[0] = *(const uint4*)&tmp[0];
            dst[1] = *(const uint4*)&tmp[8];
        }
        if (kb + 32 < K)
            __builtin_prefetch(B + (size_t)(n0 + lrow) * K + kb + 32 + lcol, 0, 0);
        if (wid == 0) tdm_wait();
        __syncthreads();

        FragB afr[4], bfr[2];
#pragma unroll
        for (int ms = 0; ms < 4; ++ms) {   // A layout: contiguous K half per lane-half
            const unsigned short* p = As + (size_t)(wm * 64 + ms * 16 + ln) * 40;
            afr[ms].u[0] = *(const uint4*)(p + half * 16);
            afr[ms].u[1] = *(const uint4*)(p + half * 16 + 8);
        }
#pragma unroll
        for (int ns = 0; ns < 2; ++ns) {   // B layout: contiguous K half per lane-half
            const unsigned short* p = Bs + (size_t)(wn * 32 + ns * 16 + ln) * 40 + half * 16;
            bfr[ns].u[0] = *(const uint4*)(p);
            bfr[ns].u[1] = *(const uint4*)(p + 8);
        }
#pragma unroll
        for (int ms = 0; ms < 4; ++ms)
#pragma unroll
            for (int ns = 0; ns < 2; ++ns)
                acc[ms][ns] = wmma_bf16(afr[ms], bfr[ns], acc[ms][ns]);
    }

#pragma unroll
    for (int ms = 0; ms < 4; ++ms)
#pragma unroll
        for (int ns = 0; ns < 2; ++ns) {
            int mbase = m0 + wm * 64 + ms * 16 + half * 8;
            int nidx  = n0 + wn * 32 + ns * 16 + ln;
#pragma unroll
            for (int v = 0; v < 8; ++v)
                C[(size_t)(mbase + v) * N + nidx] = acc[ms][ns][v];
        }
}

// ---------------- flash attention ----------------
// grid (T/64, NH), 128 threads = 4 waves; wave = 16 query rows, full D=128.
// K tile via TDM into LDS; V fragments read directly from pre-transposed V^T.
__global__ __launch_bounds__(128)
void flash_attn(const unsigned short* __restrict__ Qb,
                const unsigned short* __restrict__ Kb,
                const unsigned short* __restrict__ VbT,   // [KVH][D][T] bf16
                unsigned short* __restrict__ Ob) {
    __shared__ __align__(16) unsigned short Kt[32 * 136];     // [key][d], stride 136
    __shared__ __align__(16) unsigned short Ps[4][16 * 40];   // per-wave P scratch

    const int tid  = threadIdx.x;
    const int lane = tid & 31;
    const int wid  = tid >> 5;
    const int h    = blockIdx.y;
    const int kvh  = h >> 2;                   // rep = 4
    const int qb0  = blockIdx.x * 64;
    const int t0   = qb0 + wid * 16;
    const int ln   = lane & 15;
    const int half = lane >> 4;
    const int khalf = half * 8;

    const unsigned ktAddr = (unsigned)(uintptr_t)(&Kt[0]);
    const unsigned short* Vhead = VbT + (size_t)kvh * D_ * T_;

    // preload Q fragments (A layout), rows t0+ln, head h
    FragB qf[4];
    {
        const unsigned short* qrow = Qb + (size_t)(t0 + ln) * (NH_ * D_) + (size_t)h * D_;
#pragma unroll
        for (int f = 0; f < 4; ++f) {
            qf[f].u[0] = *(const uint4*)(qrow + f * 32 + half * 16);
            qf[f].u[1] = *(const uint4*)(qrow + f * 32 + half * 16 + 8);
        }
    }

    v8f oacc[8];
#pragma unroll
    for (int i = 0; i < 8; ++i) oacc[i] = zero8();
    float mrun[8], lrun[8];
#pragma unroll
    for (int v = 0; v < 8; ++v) { mrun[v] = NEG_BIG; lrun[v] = 0.0f; }

    const int kend = qb0 + 64;
    for (int kb = 0; kb < kend; kb += 32) {
        __syncthreads();   // protect previous iteration's readers
        if (wid == 0) {    // K tile: 32 rows x 128 bf16, row stride KVH*D
            tdm_load_2d(ktAddr,
                        Kb + (size_t)kb * (KVH_ * D_) + (size_t)kvh * D_,
                        TDM_W0_K, 128, 32, 128, 32,
                        (unsigned long long)(KVH_ * D_));
            tdm_wait();
        }
        __syncthreads();

        // S = Q K^T  (two 16x16 C tiles for 32 keys)
        v8f sA = zero8(), sB = zero8();
#pragma unroll
        for (int f = 0; f < 4; ++f) {
            FragB kf0, kf1;   // B layout: lane = key column, contiguous d half
            const unsigned short* p0 = Kt + (0 + ln) * 136 + f * 32 + half * 16;
            kf0.u[0] = *(const uint4*)(p0);
            kf0.u[1] = *(const uint4*)(p0 + 8);
            const unsigned short* p1 = Kt + (16 + ln) * 136 + f * 32 + half * 16;
            kf1.u[0] = *(const uint4*)(p1);
            kf1.u[1] = *(const uint4*)(p1 + 8);
            sA = wmma_bf16(qf[f], kf0, sA);
            sB = wmma_bf16(qf[f], kf1, sB);
        }

        // online softmax per row (8 rows per lane-half)
        unsigned short* myPs = &Ps[wid][0];
#pragma unroll
        for (int v = 0; v < 8; ++v) {
            int trow = t0 + v + half * 8;
            float a = sA[v] * SCALE_;
            float b = sB[v] * SCALE_;
            if (kb + ln      > trow) a = NEG_BIG;
            if (kb + 16 + ln > trow) b = NEG_BIG;
            float rm   = rmax16(fmaxf(a, b));
            float nm   = fmaxf(mrun[v], rm);
            float corr = __expf(mrun[v] - nm);
            float pa   = __expf(a - nm);
            float pb   = __expf(b - nm);
            float rs   = rsum16(pa + pb);
            lrun[v] = lrun[v] * corr + rs;
            mrun[v] = nm;
#pragma unroll
            for (int d8 = 0; d8 < 8; ++d8) oacc[d8][v] *= corr;
            int m = v + half * 8;
            myPs[m * 40 + ln]      = f2bf(pa);
            myPs[m * 40 + 16 + ln] = f2bf(pb);
        }
        lds_wait();   // intra-wave LDS RAW before re-layout read

        FragB pfr;    // P as A operand (16x32)
        {
            const unsigned short* pp = myPs + ln * 40;
            pfr.u[0] = *(const uint4*)(pp + half * 16);
            pfr.u[1] = *(const uint4*)(pp + half * 16 + 8);
        }
#pragma unroll
        for (int d8 = 0; d8 < 8; ++d8) {
            FragB vf;   // B operand direct from V^T: lane = d column, contiguous keys
            const unsigned short* vp = Vhead + (size_t)(d8 * 16 + ln) * T_ + kb + half * 16;
            vf.u[0] = *(const uint4*)(vp);
            vf.u[1] = *(const uint4*)(vp + 8);
            oacc[d8] = wmma_bf16(pfr, vf, oacc[d8]);
        }
    }

    // epilogue: O / l, write bf16 attn output
#pragma unroll
    for (int d8 = 0; d8 < 8; ++d8)
#pragma unroll
        for (int v = 0; v < 8; ++v) {
            int t = t0 + v + half * 8;
            int d = d8 * 16 + ln;
            float val = oacc[d8][v] / lrun[v];
            Ob[(size_t)t * (NH_ * D_) + (size_t)h * D_ + d] = f2bf(val);
        }
}

// ---------------- host orchestration ----------------
extern "C" void kernel_launch(void* const* d_in, const int* in_sizes, int n_in,
                              void* d_out, int out_size, void* d_ws, size_t ws_size,
                              hipStream_t stream) {
    (void)in_sizes; (void)n_in; (void)out_size; (void)ws_size;
    const float* X   = (const float*)d_in[0];
    const float* Wq  = (const float*)d_in[1];
    const float* Wk  = (const float*)d_in[2];
    const float* Wv  = (const float*)d_in[3];
    const float* Wo  = (const float*)d_in[4];
    const int*   pos = (const int*)d_in[5];
    float* out = (float*)d_out;

    uintptr_t base = (uintptr_t)d_ws;
    auto carve = [&](size_t bytes) -> void* {
        void* p = (void*)base;
        base += (bytes + 255) & ~(size_t)255;
        return p;
    };
    unsigned short* Xbf  = (unsigned short*)carve((size_t)T_ * HID_ * 2);
    float*          qtmp = (float*)carve((size_t)T_ * NH_ * D_ * 4);
    float*          ktmp = (float*)carve((size_t)T_ * KVH_ * D_ * 4);
    float*          vtmp = (float*)carve((size_t)T_ * KVH_ * D_ * 4);
    unsigned short* Qbf  = (unsigned short*)carve((size_t)T_ * NH_ * D_ * 2);
    unsigned short* Kbf  = (unsigned short*)carve((size_t)T_ * KVH_ * D_ * 2);
    unsigned short* VbT  = (unsigned short*)carve((size_t)KVH_ * D_ * T_ * 2);
    unsigned short* Abf  = (unsigned short*)carve((size_t)T_ * NH_ * D_ * 2);

    // 1) X -> bf16
    {
        int n = T_ * HID_;
        cvt_f32_bf16<<<(n + 255) / 256, 256, 0, stream>>>(X, Xbf, n);
    }
    // 2) projections
    gemm_bf16_f32w<<<dim3((NH_ * D_) / 128, T_ / 128), 256, 0, stream>>>(
        Xbf, Wq, qtmp, T_, NH_ * D_, HID_);
    gemm_bf16_f32w<<<dim3((KVH_ * D_) / 128, T_ / 128), 256, 0, stream>>>(
        Xbf, Wk, ktmp, T_, KVH_ * D_, HID_);
    gemm_bf16_f32w<<<dim3((KVH_ * D_) / 128, T_ / 128), 256, 0, stream>>>(
        Xbf, Wv, vtmp, T_, KVH_ * D_, HID_);
    // 3) RoPE q/k, transpose+cast v
    {
        int nq = T_ * NH_ * 64;
        rope_kernel<<<(nq + 255) / 256, 256, 0, stream>>>(qtmp, Qbf, pos, NH_, nq);
        int nk = T_ * KVH_ * 64;
        rope_kernel<<<(nk + 255) / 256, 256, 0, stream>>>(ktmp, Kbf, pos, KVH_, nk);
        transpose_cast<<<dim3((KVH_ * D_) / 32, T_ / 32), dim3(32, 8), 0, stream>>>(
            vtmp, VbT, T_, KVH_ * D_);
    }
    // 4) causal GQA flash attention
    flash_attn<<<dim3(T_ / 64, NH_), 128, 0, stream>>>(Qbf, Kbf, VbT, Abf);
    // 5) output projection
    gemm_bf16_f32w<<<dim3(HID_ / 128, T_ / 128), 256, 0, stream>>>(
        Abf, Wo, out, T_, HID_, NH_ * D_);
}
